// RNNModel_6786048328114
// MI455X (gfx1250) — matrix operand order repaired and checked by reading, not compiled, not run on hardware
//
#include <hip/hip_runtime.h>
#include <hip/hip_bf16.h>

#define DEPTH  4
#define HID    512
#define NB     64        // batch
#define SEQ    512       // T
#define G3     1536      // 3*HID
#define KC     128       // K-chunk staged per TDM transfer
#define NCHUNK (HID / KC)
#define LN_EPS 1e-5f

typedef __attribute__((ext_vector_type(16))) __bf16 v16bf;
typedef __attribute__((ext_vector_type(8)))  __bf16 v8bf;
typedef __attribute__((ext_vector_type(8)))  float  v8f;
typedef __attribute__((ext_vector_type(4)))  unsigned int v4u;
typedef __attribute__((ext_vector_type(8)))  int v8i;
typedef __attribute__((ext_vector_type(4)))  int v4i;

// ---------------------------------------------------------------------------
// Tensor Data Mover: 2D tile (tile_w x tile_h elements, 2-byte elems) from
// global (row stride `stride` elems) into LDS at byte offset lds_addr.
// Descriptor per CDNA5 ISA (D# groups 0/1).
// This toolchain exposes the 6-arg builtin:
//   (uint32x4 g0, int32x8 g1, int32x4 g2, int32x4 g3, int32x8 g4, i32 cpol)
// ---------------------------------------------------------------------------
__device__ __forceinline__ void tdm_load_2d(unsigned lds_addr, const void* gptr,
                                            unsigned tile_w, unsigned tile_h,
                                            unsigned stride,
                                            unsigned dim0, unsigned dim1) {
    unsigned long long ga = (unsigned long long)(uintptr_t)gptr;
    v4u g0;
    g0[0] = 1u;                                    // count=1, user-mode D#
    g0[1] = lds_addr;                              // lds_addr (bytes)
    g0[2] = (unsigned)(ga & 0xffffffffu);          // global_addr[31:0]
    g0[3] = (unsigned)((ga >> 32) & 0x01ffffffu)   // global_addr[56:32]
          | (2u << 30);                            // type=2 ("image")
    v8i g1;
    g1[0] = 0x00010000;                            // data_size=1 -> 2 bytes
    g1[1] = (int)((dim0 & 0xffffu) << 16);                      // tensor_dim0 lo
    g1[2] = (int)((dim0 >> 16) | ((dim1 & 0xffffu) << 16));     // dim0 hi | dim1 lo
    g1[3] = (int)((dim1 >> 16) | (tile_w << 16));               // dim1 hi | tile_dim0
    g1[4] = (int)tile_h;                                        // tile_dim1
    g1[5] = (int)stride;                                        // tensor_dim0_stride lo
    g1[6] = 0;
    g1[7] = 0;
    v4i z4  = {0, 0, 0, 0};
    v8i z8  = {0, 0, 0, 0, 0, 0, 0, 0};
    __builtin_amdgcn_tensor_load_to_lds(g0, g1, z4, z4, z8, 0);
}

__device__ __forceinline__ unsigned lds_off(const void* p) {
    // generic pointers to LDS carry the LDS byte offset in the low 32 bits
    return (unsigned)(unsigned long long)(uintptr_t)p;
}

// ---------------------------------------------------------------------------
// WMMA fragment loaders (CDNA5 wave32 layouts, CDNA5 ISA §7.12.2)
// ---------------------------------------------------------------------------
__device__ __forceinline__ v16bf load_frag_A(const __bf16* __restrict__ A,
                                             int lda, int m0, int k0, int lane) {
    int row = m0 + (lane & 15);
    int k   = k0 + ((lane >> 4) << 3);
    const __bf16* p = A + row * lda + k;
    v8bf lo = *(const v8bf*)p;          // K = k .. k+7
    v8bf hi = *(const v8bf*)(p + 16);   // K = k+16 .. k+23
    v16bf r;
#pragma unroll
    for (int i = 0; i < 8; ++i) { r[i] = lo[i]; r[i + 8] = hi[i]; }
    return r;
}

__device__ __forceinline__ v16bf load_frag_B(const __bf16* __restrict__ W,
                                             int ldb, int n0, int k0, int lane) {
    int col = n0 + (lane & 15);
    int k   = k0 + ((lane >> 4) << 3);
    const __bf16* p = W + col * ldb + k;
    v8bf lo = *(const v8bf*)p;
    v8bf hi = *(const v8bf*)(p + 16);
    v16bf r;
#pragma unroll
    for (int i = 0; i < 8; ++i) { r[i] = lo[i]; r[i + 8] = hi[i]; }
    return r;
}

// ---------------------------------------------------------------------------
// Batched GEMM with TDM double-buffered LDS staging.
// C[64 x 1536] = A[64 x 512] * W[1536 x 512]^T + bias
// grid = (24, nprob), block = 128 (4 waves). Block owns a 64(M) x 64(N) tile;
// wave w owns rows [16w, 16w+16). TDM stages A-chunk + B-chunk (64 x 128 bf16
// each) per buffer; wave 0 drives the DMA pipeline via TENSORcnt.
// ---------------------------------------------------------------------------
struct GemmArgs {
    const __bf16* A[5];
    const __bf16* W[5];
    const float*  bias[5];
    float*        C[5];
};

__global__ __launch_bounds__(128) void gemm_tdm_kernel(GemmArgs args) {
    __shared__ __bf16 sA[2][64 * KC];   // 2 x 16 KB
    __shared__ __bf16 sB[2][64 * KC];   // 2 x 16 KB

    const int lane = threadIdx.x & 31;
    const int wave = threadIdx.x >> 5;
    const int prob = blockIdx.y;
    const int n0   = blockIdx.x * 64;
    const int m0   = wave * 16;

    const __bf16* __restrict__ A  = args.A[prob];
    const __bf16* __restrict__ Wm = args.W[prob];
    const float*  __restrict__ bs = args.bias[prob];
    float*        __restrict__ C  = args.C[prob];

    if (wave == 0) {
        tdm_load_2d(lds_off(&sA[0][0]), A + 0,             KC, 64, HID, HID, NB);
        tdm_load_2d(lds_off(&sB[0][0]), Wm + n0 * HID + 0, KC, 64, HID, HID, G3);
    }

    v8f acc[4] = {v8f{}, v8f{}, v8f{}, v8f{}};

#pragma unroll
    for (int c = 0; c < NCHUNK; ++c) {
        const int buf = c & 1;
        if (wave == 0) {
            if (c + 1 < NCHUNK) {
                const int kn = (c + 1) * KC;
                tdm_load_2d(lds_off(&sA[buf ^ 1][0]), A + kn,             KC, 64, HID, HID, NB);
                tdm_load_2d(lds_off(&sB[buf ^ 1][0]), Wm + n0 * HID + kn, KC, 64, HID, HID, G3);
                __builtin_amdgcn_s_wait_tensorcnt(2);   // chunk c landed (in-order)
            } else {
                __builtin_amdgcn_s_wait_tensorcnt(0);
            }
        }
        __syncthreads();

        const __bf16* cA = &sA[buf][0];
        const __bf16* cB = &sB[buf][0];
#pragma unroll
        for (int kk = 0; kk < KC; kk += 32) {
            v16bf a = load_frag_A(cA, KC, m0, kk, lane);
#pragma unroll
            for (int j = 0; j < 4; ++j) {
                v16bf b = load_frag_B(cB, KC, 16 * j, kk, lane);
                acc[j] = __builtin_amdgcn_wmma_f32_16x16x32_bf16(
                             false, a, false, b, (short)0, acc[j], false, false);
            }
        }
        __syncthreads();   // buffer free before wave 0 refills it
    }

    // C/D layout: VGPR r : lanes 0-15 -> (m0+r, n), lanes 16-31 -> (m0+r+8, n)
    const int coll  = lane & 15;
    const int rbase = m0 + ((lane >> 4) << 3);
#pragma unroll
    for (int j = 0; j < 4; ++j) {
        const int col = n0 + 16 * j + coll;
        const float bv = bs[col];
#pragma unroll
        for (int r = 0; r < 8; ++r)
            C[(rbase + r) * G3 + col] = acc[j][r] + bv;
    }
}

// ---------------------------------------------------------------------------
// Elementwise GRU gate combine for one layer
// ---------------------------------------------------------------------------
__device__ __forceinline__ float sigmf(float x) { return 1.0f / (1.0f + __expf(-x)); }

__global__ __launch_bounds__(256) void combine_kernel(
        const float* __restrict__ gi, const float* __restrict__ gh,
        const float* __restrict__ h,  float* __restrict__ newl,
        const float* __restrict__ other_f, const __bf16* __restrict__ other_b,
        __bf16* __restrict__ a_out) {
    int i = blockIdx.x * 256 + threadIdx.x;          // 0 .. 64*512-1
    if (i >= NB * HID) return;
    int m = i >> 9, k = i & (HID - 1);
    int g = m * G3 + k;
    float r  = sigmf(gi[g]            + gh[g]);
    float z  = sigmf(gi[g + HID]      + gh[g + HID]);
    float nn = tanhf(gi[g + 2 * HID]  + r * gh[g + 2 * HID]);
    float nv = (1.0f - z) * nn + z * h[i];
    newl[i] = nv;
    if (a_out) {
        float o = other_f ? other_f[i] : (float)other_b[i];
        a_out[i] = (__bf16)(nv + o);
    }
}

// ---------------------------------------------------------------------------
// LayerNorm over each (layer,batch) row of 512; writes fp32 h and bf16 h copy
// ---------------------------------------------------------------------------
__global__ __launch_bounds__(256) void ln_kernel(
        const float* __restrict__ newh, const float* __restrict__ g,
        const float* __restrict__ bta,  float* __restrict__ hF,
        __bf16* __restrict__ hB) {
    __shared__ float s_sum[256];
    __shared__ float s_sq[256];
    const int row = blockIdx.x;                      // 0 .. DEPTH*NB-1
    const int t   = threadIdx.x;
    const float* x = newh + row * HID;
    float v0 = x[t], v1 = x[t + 256];
    s_sum[t] = v0 + v1;
    s_sq[t]  = v0 * v0 + v1 * v1;
    __syncthreads();
#pragma unroll
    for (int off = 128; off > 0; off >>= 1) {
        if (t < off) { s_sum[t] += s_sum[t + off]; s_sq[t] += s_sq[t + off]; }
        __syncthreads();
    }
    const float mean = s_sum[0] * (1.0f / HID);
    const float var  = s_sq[0] * (1.0f / HID) - mean * mean;
    const float rstd = rsqrtf(var + LN_EPS);
    const int base = row * HID;
    float o0 = (v0 - mean) * rstd * g[t]       + bta[t];
    float o1 = (v1 - mean) * rstd * g[t + 256] + bta[t + 256];
    hF[base + t]       = o0;  hB[base + t]       = (__bf16)o0;
    hF[base + t + 256] = o1;  hB[base + t + 256] = (__bf16)o1;
}

// ---------------------------------------------------------------------------
// Setup kernels
// ---------------------------------------------------------------------------
__global__ __launch_bounds__(256) void cvt_weights_kernel(
        const float* __restrict__ Wih, const float* __restrict__ Whh,
        __bf16* __restrict__ WihB, __bf16* __restrict__ WhhB) {
    int i = blockIdx.x * 256 + threadIdx.x;
    if (i < DEPTH * G3 * HID) {
        WihB[i] = (__bf16)Wih[i];
        WhhB[i] = (__bf16)Whh[i];
    }
}

__global__ __launch_bounds__(256) void embed_kernel(
        const int* __restrict__ x, const float* __restrict__ emb,
        __bf16* __restrict__ Xe) {
    int i = blockIdx.x * 256 + threadIdx.x;          // T*B*HID = 16.7M
    if (i >= SEQ * NB * HID) return;
    int e  = i & (HID - 1);
    int tb = i >> 9;
    int t  = tb >> 6;
    int b  = tb & 63;
    int tok = x[b * SEQ + t];
    Xe[i] = (__bf16)emb[(long)tok * HID + e];
}

__global__ __launch_bounds__(256) void zero_h_kernel(float* hF, __bf16* hB) {
    int i = blockIdx.x * 256 + threadIdx.x;
    if (i < DEPTH * NB * HID) { hF[i] = 0.0f; hB[i] = (__bf16)0.0f; }
}

// ---------------------------------------------------------------------------
// Final FC: out[b][c] = h3[b]·fc_w[c] + fc_b[c]   (64 x 5)
// ---------------------------------------------------------------------------
__global__ __launch_bounds__(32) void fc_kernel(
        const float* __restrict__ h3, const float* __restrict__ fcw,
        const float* __restrict__ fcb, float* __restrict__ out) {
    const int b = blockIdx.x, lane = threadIdx.x;
#pragma unroll
    for (int c = 0; c < 5; ++c) {
        float s = 0.0f;
        for (int k = lane; k < HID; k += 32)
            s += h3[b * HID + k] * fcw[c * HID + k];
#pragma unroll
        for (int off = 16; off > 0; off >>= 1) s += __shfl_xor(s, off, 32);
        if (lane == 0) out[b * 5 + c] = s + fcb[c];
    }
}

// ---------------------------------------------------------------------------
extern "C" void kernel_launch(void* const* d_in, const int* in_sizes, int n_in,
                              void* d_out, int out_size, void* d_ws, size_t ws_size,
                              hipStream_t stream) {
    const int*   x     = (const int*)  d_in[0];
    const float* emb   = (const float*)d_in[1];
    const float* W_ih  = (const float*)d_in[2];
    const float* W_hh  = (const float*)d_in[3];
    const float* b_ih  = (const float*)d_in[4];
    const float* b_hh  = (const float*)d_in[5];
    const float* ln_g  = (const float*)d_in[6];
    const float* ln_b  = (const float*)d_in[7];
    const float* fc_w  = (const float*)d_in[8];
    const float* fc_b  = (const float*)d_in[9];
    float* out = (float*)d_out;

    // ---- workspace carve-up (all 256B aligned) ----
    size_t off = 0;
    auto carve = [&](size_t bytes) -> void* {
        void* p = (char*)d_ws + off;
        off += (bytes + 255) & ~(size_t)255;
        return p;
    };
    __bf16* WihB = (__bf16*)carve((size_t)DEPTH * G3 * HID * 2);
    __bf16* WhhB = (__bf16*)carve((size_t)DEPTH * G3 * HID * 2);
    __bf16* Xe   = (__bf16*)carve((size_t)SEQ * NB * HID * 2);
    float*  hF   = (float*) carve((size_t)DEPTH * NB * HID * 4);
    __bf16* hB   = (__bf16*)carve((size_t)DEPTH * NB * HID * 2);
    float*  newF = (float*) carve((size_t)DEPTH * NB * HID * 4);
    float*  gh   = (float*) carve((size_t)DEPTH * NB * G3 * 4);
    float*  gi0  = (float*) carve((size_t)NB * G3 * 4);
    float*  gic  = (float*) carve((size_t)NB * G3 * 4);
    __bf16* a1   = (__bf16*)carve((size_t)NB * HID * 2);
    __bf16* a2   = (__bf16*)carve((size_t)NB * HID * 2);
    __bf16* a3   = (__bf16*)carve((size_t)NB * HID * 2);
    (void)ws_size; (void)n_in; (void)in_sizes; (void)out_size;

    // ---- one-time setup ----
    cvt_weights_kernel<<<(DEPTH * G3 * HID + 255) / 256, 256, 0, stream>>>(
        W_ih, W_hh, WihB, WhhB);
    embed_kernel<<<(SEQ * NB * HID + 255) / 256, 256, 0, stream>>>(x, emb, Xe);
    zero_h_kernel<<<(DEPTH * NB * HID + 255) / 256, 256, 0, stream>>>(hF, hB);

    const int CB = (NB * HID + 255) / 256;   // combine grid

    for (int t = 0; t < SEQ; ++t) {
        const __bf16* Xt = Xe + (size_t)t * NB * HID;

        // K1: parallel batch {gi0, gh0..gh3}
        GemmArgs g1;
        g1.A[0] = Xt;  g1.W[0] = WihB;  g1.bias[0] = b_ih;  g1.C[0] = gi0;
        for (int l = 0; l < DEPTH; ++l) {
            g1.A[1 + l]    = hB   + (size_t)l * NB * HID;
            g1.W[1 + l]    = WhhB + (size_t)l * G3 * HID;
            g1.bias[1 + l] = b_hh + (size_t)l * G3;
            g1.C[1 + l]    = gh   + (size_t)l * NB * G3;
        }
        gemm_tdm_kernel<<<dim3(24, 5), 128, 0, stream>>>(g1);

        // layer 0 combine : new0 ; a1 = new0 + x_t
        combine_kernel<<<CB, 256, 0, stream>>>(gi0, gh, hF, newF,
                                               nullptr, Xt, a1);

        // layers 1..3 : gemm(gi) -> combine
        const __bf16* ain[3]  = {a1, a2, a3};
        __bf16*       aout[3] = {a2, a3, nullptr};
        for (int l = 1; l < DEPTH; ++l) {
            GemmArgs gl;
            gl.A[0]    = ain[l - 1];
            gl.W[0]    = WihB + (size_t)l * G3 * HID;
            gl.bias[0] = b_ih + (size_t)l * G3;
            gl.C[0]    = gic;
            for (int p = 1; p < 5; ++p) { gl.A[p] = gl.A[0]; gl.W[p] = gl.W[0];
                                          gl.bias[p] = gl.bias[0]; gl.C[p] = gl.C[0]; }
            gemm_tdm_kernel<<<dim3(24, 1), 128, 0, stream>>>(gl);

            combine_kernel<<<CB, 256, 0, stream>>>(
                gic, gh + (size_t)l * NB * G3, hF + (size_t)l * NB * HID,
                newF + (size_t)l * NB * HID,
                newF + (size_t)(l - 1) * NB * HID, nullptr, aout[l - 1]);
        }

        // LayerNorm all DEPTH*NB rows -> h(t) fp32 + bf16
        ln_kernel<<<DEPTH * NB, 256, 0, stream>>>(newF, ln_g, ln_b, hF, hB);
    }

    // final classifier on h[3]
    fc_kernel<<<NB, 32, 0, stream>>>(hF + (size_t)3 * NB * HID, fc_w, fc_b, out);
}
